// HGCN_87351044866138
// MI455X (gfx1250) — matrix operand order repaired and checked by reading, not compiled, not run on hardware
//
#include <hip/hip_runtime.h>
#include <hip/hip_bf16.h>
#include <math.h>

typedef __attribute__((ext_vector_type(16))) _Float16 v16h;
typedef __attribute__((ext_vector_type(8)))  _Float16 v8h;
typedef __attribute__((ext_vector_type(8)))  float    v8f;

#define HID 128
#define IN_DIM 256

// ---------------------------------------------------------------- utilities
__global__ void fill_f32(float* __restrict__ p, float v, int n) {
  int i = blockIdx.x * blockDim.x + threadIdx.x;
  if (i < n) p[i] = v;
}

__global__ void deg_count(const int* __restrict__ col, int E, float* __restrict__ deg) {
  int i = blockIdx.x * blockDim.x + threadIdx.x;
  if (i < E) atomicAdd(&deg[col[i]], 1.0f);
}

__global__ void finalize_dinv(float* __restrict__ deg, int n) {
  int i = blockIdx.x * blockDim.x + threadIdx.x;
  if (i < n) { float d = deg[i]; deg[i] = (d > 0.0f) ? rsqrtf(d) : 0.0f; }
}

__global__ void cvt_f32_f16(const float* __restrict__ s, _Float16* __restrict__ d, int n) {
  int i = blockIdx.x * blockDim.x + threadIdx.x;
  if (i < n) d[i] = (_Float16)s[i];
}

// Convert f32 weight W[K,128] into f16 WMMA-B fragment order:
// Bp[((kt*8 + nt)*32 + lane)*16 + i] = (f16) W[kt*32 + (lane>>4)*16 + i][nt*16 + (lane&15)]
// One thread per (kt, nt, lane) gathers its 16 halves (32 contiguous output bytes).
__global__ void pack_W_f16(const float* __restrict__ W, _Float16* __restrict__ Bp, int K) {
  int idx = blockIdx.x * blockDim.x + threadIdx.x;
  int total = (K >> 5) * 8 * 32;
  if (idx >= total) return;
  int lane = idx & 31;
  int nt   = (idx >> 5) & 7;
  int kt   = idx >> 8;
  int m  = lane & 15;
  int hi = lane >> 4;
  const float* src = W + (size_t)(kt * 32 + hi * 16) * HID + nt * 16 + m;
  _Float16* dst = Bp + (size_t)idx * 16;
#pragma unroll
  for (int i = 0; i < 16; ++i) dst[i] = (_Float16)src[(size_t)i * HID];
}

// ---------------------------------------------------------------- WMMA GEMM
// C[M,128] = A[M,K] (f16 row-major) * W (f16, pre-packed fragments), f32 accum.
// grid.x = M/16; 256 threads = 8 waves, one 16x16 N-tile per wave.
// A stripe (16 x K halves) staged in LDS, shared by all 8 waves.
__global__ void __launch_bounds__(256)
wmma_gemm_f16(const _Float16* __restrict__ A, const _Float16* __restrict__ Bp,
              float* __restrict__ C, int K) {
  extern __shared__ float4 smem[];
  _Float16* As = (_Float16*)smem;          // [16][K] halves

  const int row0 = blockIdx.x * 16;
  const int wave = threadIdx.x >> 5;       // 0..7 -> N tile
  const int lane = threadIdx.x & 31;
  const int m    = lane & 15;
  const int hi   = lane >> 4;

  // cooperative load of the 16-row A stripe (8 halves = 16B per slot)
  const int slots = 16 * (K >> 3);         // 16B chunks
  for (int t = threadIdx.x; t < slots; t += 256) {
    int r  = t / (K >> 3);
    int c8 = t - r * (K >> 3);
    *(v8h*)&As[r * K + c8 * 8] = *(const v8h*)&A[(size_t)(row0 + r) * K + c8 * 8];
  }
  __syncthreads();

  v8f acc = {};
  const v8h* bfrag = (const v8h*)(Bp + ((size_t)(wave * 32 + lane)) * 16);
  const int bstep = (8 * 32 * 16) / 8;     // v8h stride per kTile
  for (int kb = 0; kb < K; kb += 32) {
    // A fragment from LDS: two contiguous 16B chunks per lane
    const _Float16* arow = As + m * K + kb + hi * 16;
    v8h alo = *(const v8h*)arow;
    v8h ahi = *(const v8h*)(arow + 8);
    v16h a = __builtin_shufflevector(alo, ahi, 0,1,2,3,4,5,6,7,8,9,10,11,12,13,14,15);
    // B fragment: 32 contiguous bytes per lane from the packed buffer
    v8h blo = bfrag[0];
    v8h bhi = bfrag[1];
    v16h b = __builtin_shufflevector(blo, bhi, 0,1,2,3,4,5,6,7,8,9,10,11,12,13,14,15);
    bfrag += bstep;

    acc = __builtin_amdgcn_wmma_f32_16x16x32_f16(
        false, a, false, b, (short)0, acc, false, false);
  }
  // C fragment: elem v -> row row0 + v + hi*8, col wave*16 + m.
  float* cp = C + (size_t)row0 * HID + wave * 16 + m;
#pragma unroll
  for (int v = 0; v < 8; ++v) cp[(size_t)(v + hi * 8) * HID] = acc[v];
}

// ------------------------------------------------------- aggregation kernels
// agg[i,k] = lin[i,k] * dinv[i]^2   (self-loop contribution, also the init)
__global__ void init_agg(const float* __restrict__ lin, const float* __restrict__ dinv,
                         float* __restrict__ agg, int total) {
  int idx = blockIdx.x * blockDim.x + threadIdx.x;
  if (idx >= total) return;
  float di = dinv[idx >> 7];
  agg[idx] = lin[idx] * di * di;
}

// one wave per edge: agg[col] += lin[row] * dinv[row]*dinv[col]
// lane owns channels 4L..4L+3: one b128 load + 4 f32 atomics.
__global__ void scatter_edges(const float* __restrict__ lin, const float* __restrict__ dinv,
                              const int* __restrict__ row, const int* __restrict__ col,
                              int E, float* __restrict__ agg) {
  int e = blockIdx.x * (blockDim.x >> 5) + (threadIdx.x >> 5);
  if (e >= E) return;
  int lane = threadIdx.x & 31;
  int r = row[e], c = col[e];
  float nrm = dinv[r] * dinv[c];
  int k4 = lane * 4;
  const float4 v = *(const float4*)(lin + (size_t)r * HID + k4);
  float* dst = agg + (size_t)c * HID + k4;
  atomicAdd(dst + 0, v.x * nrm);
  atomicAdd(dst + 1, v.y * nrm);
  atomicAdd(dst + 2, v.z * nrm);
  atomicAdd(dst + 3, v.w * nrm);
}

// h = relu(0.5*((aggP + bP) + (aggG + bG))); write f16 (next GEMM input) and
// optionally f32 (final z straight into d_out).
__global__ void combine_relu(const float* __restrict__ aggP, const float* __restrict__ bP,
                             const float* __restrict__ aggG, const float* __restrict__ bG,
                             _Float16* __restrict__ act16, float* __restrict__ zout,
                             int total) {
  int idx = blockIdx.x * blockDim.x + threadIdx.x;
  if (idx >= total) return;
  int k = idx & (HID - 1);
  float v = 0.5f * ((aggP[idx] + bP[k]) + (aggG[idx] + bG[k]));
  v = fmaxf(v, 0.0f);
  act16[idx] = (_Float16)v;
  if (zout) zout[idx] = v;
}

// --------------------------------------------------------------- classifier
__global__ void classify(const float* __restrict__ z, const int* __restrict__ src,
                         const int* __restrict__ dst, const float* __restrict__ Wc,
                         const float* __restrict__ bc, float* __restrict__ probs, int P) {
  int p = blockIdx.x * blockDim.x + threadIdx.x;
  if (p >= P) return;
  const float4* zs = (const float4*)(z + (size_t)src[p] * HID);
  const float4* zd = (const float4*)(z + (size_t)dst[p] * HID);
  float a0 = bc[0], a1 = bc[1];
#pragma unroll 8
  for (int q = 0; q < HID / 4; ++q) {
    float4 v = zs[q];
    int k = q * 4;
    a0 += v.x * Wc[(k+0)*2+0] + v.y * Wc[(k+1)*2+0] + v.z * Wc[(k+2)*2+0] + v.w * Wc[(k+3)*2+0];
    a1 += v.x * Wc[(k+0)*2+1] + v.y * Wc[(k+1)*2+1] + v.z * Wc[(k+2)*2+1] + v.w * Wc[(k+3)*2+1];
  }
#pragma unroll 8
  for (int q = 0; q < HID / 4; ++q) {
    float4 v = zd[q];
    int k = HID + q * 4;
    a0 += v.x * Wc[(k+0)*2+0] + v.y * Wc[(k+1)*2+0] + v.z * Wc[(k+2)*2+0] + v.w * Wc[(k+3)*2+0];
    a1 += v.x * Wc[(k+0)*2+1] + v.y * Wc[(k+1)*2+1] + v.z * Wc[(k+2)*2+1] + v.w * Wc[(k+3)*2+1];
  }
  probs[(size_t)p * 2 + 0] = 1.0f / (1.0f + expf(-a0));
  probs[(size_t)p * 2 + 1] = 1.0f / (1.0f + expf(-a1));
}

// ------------------------------------------------------------------- driver
extern "C" void kernel_launch(void* const* d_in, const int* in_sizes, int n_in,
                              void* d_out, int out_size, void* d_ws, size_t ws_size,
                              hipStream_t stream) {
  const float* x      = (const float*)d_in[0];
  const int*   ei_ppi = (const int*)d_in[1];
  const int*   ei_go  = (const int*)d_in[2];
  const int*   cand   = (const int*)d_in[3];
  const float* W0p = (const float*)d_in[4];
  const float* b0p = (const float*)d_in[5];
  const float* W0g = (const float*)d_in[6];
  const float* b0g = (const float*)d_in[7];
  const float* W1p = (const float*)d_in[8];
  const float* b1p = (const float*)d_in[9];
  const float* W1g = (const float*)d_in[10];
  const float* b1g = (const float*)d_in[11];
  const float* Wc  = (const float*)d_in[12];
  const float* bc  = (const float*)d_in[13];

  const int N  = in_sizes[0] / IN_DIM;   // 50000 (multiple of 16)
  const int Ep = in_sizes[1] / 2;
  const int Eg = in_sizes[2] / 2;
  const int P  = in_sizes[3] / 2;
  const int tot = N * HID;

  float* zout  = (float*)d_out;              // [N,128]
  float* probs = zout + (size_t)N * HID;     // [P,2]

  // carve workspace (~116 MB)
  char* w = (char*)d_ws;
  auto take = [&](size_t bytes) -> char* {
    char* p = w; w += (bytes + 255) & ~(size_t)255; return p;
  };
  float*    dinvP = (float*)   take((size_t)N * 4);
  float*    dinvG = (float*)   take((size_t)N * 4);
  _Float16* x16   = (_Float16*)take((size_t)N * IN_DIM * 2);
  _Float16* act16 = (_Float16*)take((size_t)N * HID * 2);
  _Float16* wpk   = (_Float16*)take((size_t)IN_DIM * HID * 2);  // packed weight fragments
  float*    lin   = (float*)   take((size_t)tot * 4);
  float*    aggP  = (float*)   take((size_t)tot * 4);
  float*    aggG  = (float*)   take((size_t)tot * 4);

  dim3 b256(256);
  auto g1 = [](int n) { return dim3((unsigned)((n + 255) / 256)); };
  dim3 gGemm((unsigned)(N / 16));
  dim3 gScatP((unsigned)((Ep + 7) / 8)), gScatG((unsigned)((Eg + 7) / 8));
  const size_t lds0 = (size_t)16 * IN_DIM * 2;   // A stripe, layer 0 (8 KB)
  const size_t lds1 = (size_t)16 * HID * 2;      // A stripe, layer 1 (4 KB)
  const int nFragP0 = (IN_DIM >> 5) * 8 * 32;    // pack threads, K=256
  const int nFragP1 = (HID   >> 5) * 8 * 32;     // pack threads, K=128

  // degrees (self-loop => start at 1) -> dinv
  fill_f32<<<g1(N), b256, 0, stream>>>(dinvP, 1.0f, N);
  fill_f32<<<g1(N), b256, 0, stream>>>(dinvG, 1.0f, N);
  deg_count<<<g1(Ep), b256, 0, stream>>>(ei_ppi + Ep, Ep, dinvP);
  deg_count<<<g1(Eg), b256, 0, stream>>>(ei_go + Eg, Eg, dinvG);
  finalize_dinv<<<g1(N), b256, 0, stream>>>(dinvP, N);
  finalize_dinv<<<g1(N), b256, 0, stream>>>(dinvG, N);

  cvt_f32_f16<<<g1(N * IN_DIM), b256, 0, stream>>>(x, x16, N * IN_DIM);

  // ---------------- layer 0 ----------------
  pack_W_f16<<<g1(nFragP0), b256, 0, stream>>>(W0p, wpk, IN_DIM);
  wmma_gemm_f16<<<gGemm, b256, lds0, stream>>>(x16, wpk, lin, IN_DIM);
  init_agg<<<g1(tot), b256, 0, stream>>>(lin, dinvP, aggP, tot);
  scatter_edges<<<gScatP, b256, 0, stream>>>(lin, dinvP, ei_ppi, ei_ppi + Ep, Ep, aggP);

  pack_W_f16<<<g1(nFragP0), b256, 0, stream>>>(W0g, wpk, IN_DIM);
  wmma_gemm_f16<<<gGemm, b256, lds0, stream>>>(x16, wpk, lin, IN_DIM);
  init_agg<<<g1(tot), b256, 0, stream>>>(lin, dinvG, aggG, tot);
  scatter_edges<<<gScatG, b256, 0, stream>>>(lin, dinvG, ei_go, ei_go + Eg, Eg, aggG);

  combine_relu<<<g1(tot), b256, 0, stream>>>(aggP, b0p, aggG, b0g, act16, nullptr, tot);

  // ---------------- layer 1 ----------------
  pack_W_f16<<<g1(nFragP1), b256, 0, stream>>>(W1p, wpk, HID);
  wmma_gemm_f16<<<gGemm, b256, lds1, stream>>>(act16, wpk, lin, HID);
  init_agg<<<g1(tot), b256, 0, stream>>>(lin, dinvP, aggP, tot);
  scatter_edges<<<gScatP, b256, 0, stream>>>(lin, dinvP, ei_ppi, ei_ppi + Ep, Ep, aggP);

  pack_W_f16<<<g1(nFragP1), b256, 0, stream>>>(W1g, wpk, HID);
  wmma_gemm_f16<<<gGemm, b256, lds1, stream>>>(act16, wpk, lin, HID);
  init_agg<<<g1(tot), b256, 0, stream>>>(lin, dinvG, aggG, tot);
  scatter_edges<<<gScatG, b256, 0, stream>>>(lin, dinvG, ei_go, ei_go + Eg, Eg, aggG);

  combine_relu<<<g1(tot), b256, 0, stream>>>(aggP, b1p, aggG, b1g, act16, zout, tot);

  // ---------------- classifier ----------------
  classify<<<g1(P), b256, 0, stream>>>(zout, cand, cand + P, Wc, bc, probs, P);
}
